// MultiHeadAttention_21818433864446
// MI455X (gfx1250) — matrix operand order repaired and checked by reading, not compiled, not run on hardware
//
#include <hip/hip_runtime.h>

#define BATCH   4
#define NH      16
#define SEQ     2048
#define DMODEL  1024
#define HD      64
#define ATT_SCALE 0.125f   // 1/sqrt(64)

typedef __attribute__((ext_vector_type(16))) __bf16 bf16x16;
typedef __attribute__((ext_vector_type(8)))  __bf16 bf16x8;
typedef __attribute__((ext_vector_type(4)))  __bf16 bf16x4;
typedef __attribute__((ext_vector_type(8)))  float  f32x8;

#if defined(__AMDGCN__) && \
    __has_builtin(__builtin_amdgcn_global_load_async_to_lds_b128) && \
    __has_builtin(__builtin_amdgcn_s_wait_asynccnt)
#define USE_ASYNC 1
#else
#define USE_ASYNC 0
#endif

// A-operand (16x32 bf16): lane L holds row M=(L&15), K in {koff..koff+7, 16+koff..23+koff},
// koff = 8*(L>>4).  'lo' points at K=koff, 'hi' at K=16+koff. Two 16B loads.
static __device__ __forceinline__ bf16x16 load_a_tile(const __bf16* lo, const __bf16* hi) {
  bf16x8 l = *(const bf16x8*)lo;
  bf16x8 h = *(const bf16x8*)hi;
  bf16x16 r;
#pragma unroll
  for (int i = 0; i < 8; ++i) { r[i] = l[i]; r[i + 8] = h[i]; }
  return r;
}

// B-operand (32x16 bf16): lane holds column N, K in {koff..koff+7, 16+koff..23+koff},
// koff = 8*(lane>>4), in an [N][K]-major source. Two 16B loads.
static __device__ __forceinline__ bf16x16 load_b_tile(const __bf16* p) {
  bf16x8 l = *(const bf16x8*)p;
  bf16x8 h = *(const bf16x8*)(p + 16);
  bf16x16 r;
#pragma unroll
  for (int i = 0; i < 8; ++i) { r[i] = l[i]; r[i + 8] = h[i]; }
  return r;
}

#if USE_ASYNC
typedef int v4i_vs __attribute__((vector_size(16)));
typedef __attribute__((address_space(1))) v4i_vs gv4i;   // global (AS1 / __device__)
typedef __attribute__((address_space(3))) v4i_vs lv4i;   // LDS   (AS3 / __shared__)

// one async 16B global->LDS copy (ASYNCcnt path, no VGPR round-trip)
static __device__ __forceinline__ void async_cp16(__bf16* lds, const __bf16* g) {
  __builtin_amdgcn_global_load_async_to_lds_b128((gv4i*)g, (lv4i*)lds, 0, 0);
}
#endif

// ---------------------------------------------------------------------------
// One-shot: Wt[n][k] = bf16(W[k][n]).  3 x 2MB output; cost ~= memcpy.
// ---------------------------------------------------------------------------
__global__ __launch_bounds__(256) void mha_wt_bf16(const float* __restrict__ W,
                                                   __bf16* __restrict__ Wt) {
  __shared__ __bf16 T[64][72];                // pad to dodge bank conflicts
  const int tid = threadIdx.x;
  const int kb = blockIdx.x * 64, nb = blockIdx.y * 64;
#pragma unroll
  for (int e = 0; e < 16; ++e) {
    int idx = e * 256 + tid;
    int k = idx >> 6, n = idx & 63;           // coalesced along n
    T[k][n] = (__bf16)W[(size_t)(kb + k) * DMODEL + nb + n];
  }
  __syncthreads();
#pragma unroll
  for (int e = 0; e < 16; ++e) {
    int idx = e * 256 + tid;
    int n = idx >> 6, k = idx & 63;           // coalesced along k
    Wt[(size_t)(nb + n) * DMODEL + kb + k] = T[k][n];
  }
}

// ---------------------------------------------------------------------------
// Projection GEMM: 4 waves, 64x128 output tile, K chunked by 32.
// Wt is bf16 [n][k] so B staging is pure vector copy.  vtrans=1 stores the
// head tiles dim-major ([b][h][d][s]) for the attention V operand.
// ---------------------------------------------------------------------------
__global__ __launch_bounds__(128) void mha_proj_wmma(const float* __restrict__ X,
                                                     const __bf16* __restrict__ Wt,
                                                     const float* __restrict__ bias,
                                                     __bf16* __restrict__ dst,
                                                     int vtrans) {
  __shared__ __align__(16) __bf16 Xs[64 * 32];    // [row][k]
  __shared__ __align__(16) __bf16 Ws[128 * 32];   // [col][k]
  const int tid  = threadIdx.x;
  const int wave = tid >> 5, lane = tid & 31, ln = lane & 15, lh = lane >> 4;
  const int mbase = blockIdx.x * 64;
  const int nbase = blockIdx.y * 128;

  f32x8 c[8] = {};

  for (int kb = 0; kb < DMODEL; kb += 32) {
    { // X chunk: thread owns 16 contiguous floats of one row
      int row = tid >> 1, seg = (tid & 1) * 16;
      const float* src = &X[(size_t)(mbase + row) * DMODEL + kb + seg];
      float4 f0 = *(const float4*)(src);
      float4 f1 = *(const float4*)(src + 4);
      float4 f2 = *(const float4*)(src + 8);
      float4 f3 = *(const float4*)(src + 12);
      bf16x8 lo, hi;
      lo[0]=(__bf16)f0.x; lo[1]=(__bf16)f0.y; lo[2]=(__bf16)f0.z; lo[3]=(__bf16)f0.w;
      lo[4]=(__bf16)f1.x; lo[5]=(__bf16)f1.y; lo[6]=(__bf16)f1.z; lo[7]=(__bf16)f1.w;
      hi[0]=(__bf16)f2.x; hi[1]=(__bf16)f2.y; hi[2]=(__bf16)f2.z; hi[3]=(__bf16)f2.w;
      hi[4]=(__bf16)f3.x; hi[5]=(__bf16)f3.y; hi[6]=(__bf16)f3.z; hi[7]=(__bf16)f3.w;
      *(bf16x8*)&Xs[row * 32 + seg]     = lo;
      *(bf16x8*)&Xs[row * 32 + seg + 8] = hi;
    }
#pragma unroll
    for (int e = 0; e < 4; ++e) {  // W chunk: 128x32 bf16, pure vector copy
      int idx = e * 128 + tid;
      int n = idx >> 2, k8 = (idx & 3) * 8;
      *(bf16x8*)&Ws[n * 32 + k8] =
          *(const bf16x8*)&Wt[(size_t)(nbase + n) * DMODEL + kb + k8];
    }
    __syncthreads();

    const __bf16* ar = &Xs[(wave * 16 + ln) * 32 + lh * 8];
    bf16x16 a = load_a_tile(ar, ar + 16);
#pragma unroll
    for (int nt = 0; nt < 8; ++nt) {
      bf16x16 bm = load_b_tile(&Ws[(nt * 16 + ln) * 32 + lh * 8]);
      c[nt] = __builtin_amdgcn_wmma_f32_16x16x32_bf16(false, a, false, bm,
                                                      (short)0, c[nt], false, false);
    }
    __syncthreads();
  }

#pragma unroll
  for (int nt = 0; nt < 8; ++nt) {
    int col = nbase + nt * 16 + ln;
    int h = col >> 6, d = col & 63;
    float bv = bias[col];
#pragma unroll
    for (int r = 0; r < 8; ++r) {
      int m = mbase + wave * 16 + lh * 8 + r;       // C layout: M = r + 8*laneHalf
      int b = m >> 11, s = m & (SEQ - 1);
      size_t off = vtrans ? ((((size_t)b * NH + h) * HD + d) * SEQ + s)
                          : ((((size_t)b * NH + h) * SEQ + s) * HD + d);
      dst[off] = (__bf16)(c[nt][r] + bv);
    }
  }
}

// ---------------------------------------------------------------------------
// Flash attention: 4 waves, 64 q-rows of one (b,h); 64-key chunks.
// Score tile t covers keys {4n+t} so each lane's 4 exp values are contiguous
// Pb columns -> single packed b64 store per row; Pb stays [row][key] order.
// V is consumed dim-major so its LDS tile is a straight vector copy.
// When available, K/V tiles stream via GLOBAL_LOAD_ASYNC_TO_LDS_B128 into
// double buffers so chunk i+1's fetch overlaps chunk i's 16 WMMAs.
// ---------------------------------------------------------------------------
__global__ __launch_bounds__(128) void mha_flash_wmma(const __bf16* __restrict__ qws,
                                                      const __bf16* __restrict__ kws,
                                                      const __bf16* __restrict__ vwsT,
                                                      const int* __restrict__ mask,
                                                      float* __restrict__ out) {
#if USE_ASYNC
  __shared__ __align__(16) __bf16 Ks[2][64 * 64];   // [key][d]
  __shared__ __align__(16) __bf16 Vt[2][64 * 64];   // [d][key]
#else
  __shared__ __align__(16) __bf16 Ks[1][64 * 64];
  __shared__ __align__(16) __bf16 Vt[1][64 * 64];
#endif
  __shared__ __align__(16) __bf16 Pb[4][16 * 64];   // per-wave P [row][key]
  const int tid  = threadIdx.x;
  const int wave = tid >> 5, lane = tid & 31, ln = lane & 15, lh = lane >> 4;
  const int qt = blockIdx.x, h = blockIdx.y, b = blockIdx.z;

  const __bf16* qh = qws  + (((size_t)b * NH + h) * SEQ) * HD;
  const __bf16* kh = kws  + (((size_t)b * NH + h) * SEQ) * HD;
  const __bf16* vh = vwsT + (((size_t)b * NH + h) * HD) * SEQ;  // [d][s]
  const int* mrow = mask + b * SEQ;
  const int qrow = qt * 64 + wave * 16 + ln;

  bf16x16 aq0 = load_a_tile(qh + (size_t)qrow * HD +      lh * 8,
                            qh + (size_t)qrow * HD + 16 + lh * 8);
  bf16x16 aq1 = load_a_tile(qh + (size_t)qrow * HD + 32 + lh * 8,
                            qh + (size_t)qrow * HD + 48 + lh * 8);

  f32x8 acc[4] = {};
  float vm[8], vl[8];
#pragma unroll
  for (int r = 0; r < 8; ++r) { vm[r] = -INFINITY; vl[r] = 0.f; }

  // per-thread tile-fetch coordinates (8 x 16B per chunk)
  const int fkey = tid >> 3, fd8 = (tid & 7) * 8;   // K tile element
  const int fdim = tid >> 3, fk8 = (tid & 7) * 8;   // V tile element

#if USE_ASYNC
  { // prologue: prefetch chunk 0 into buffer 0
#pragma unroll
    for (int e = 0; e < 4; ++e) {
      async_cp16(&Ks[0][(fkey + e * 16) * 64 + fd8], &kh[(size_t)(fkey + e * 16) * HD + fd8]);
      async_cp16(&Vt[0][(fdim + e * 16) * 64 + fk8], &vh[(size_t)(fdim + e * 16) * SEQ + fk8]);
    }
  }
#endif

  for (int kc = 0; kc < SEQ; kc += 64) {
#if USE_ASYNC
    const int buf = (kc >> 6) & 1;
    __syncthreads();                         // all waves done reading buf^1
    if (kc + 64 < SEQ) {                     // prefetch next chunk into buf^1
      const int nkc = kc + 64;
#pragma unroll
      for (int e = 0; e < 4; ++e) {
        async_cp16(&Ks[buf ^ 1][(fkey + e * 16) * 64 + fd8],
                   &kh[(size_t)(nkc + fkey + e * 16) * HD + fd8]);
        async_cp16(&Vt[buf ^ 1][(fdim + e * 16) * 64 + fk8],
                   &vh[(size_t)(fdim + e * 16) * SEQ + nkc + fk8]);
      }
      __builtin_amdgcn_s_wait_asynccnt(8);   // own current-chunk loads done
    } else {
      __builtin_amdgcn_s_wait_asynccnt(0);
    }
    __syncthreads();                         // publish current buffer
#else
    const int buf = 0;
    __syncthreads();                         // protect tiles read last iter
#pragma unroll
    for (int e = 0; e < 4; ++e) {            // K tile [key][d], vector copy
      int key = fkey + e * 16;
      *(bf16x8*)&Ks[0][key * 64 + fd8] = *(const bf16x8*)&kh[(size_t)(kc + key) * HD + fd8];
    }
#pragma unroll
    for (int e = 0; e < 4; ++e) {            // V tile [d][key], vector copy
      int d = fdim + e * 16;
      *(bf16x8*)&Vt[0][d * 64 + fk8] = *(const bf16x8*)&vh[(size_t)d * SEQ + kc + fk8];
    }
    __syncthreads();
#endif

    // scores: tile t covers keys {4n+t}; lane column n=ln -> key 4*ln+t
    f32x8 s[4] = {};
#pragma unroll
    for (int t = 0; t < 4; ++t) {
      const __bf16* kr = &Ks[buf][(4 * ln + t) * 64 + lh * 8];
      s[t] = __builtin_amdgcn_wmma_f32_16x16x32_bf16(false, aq0, false,
               load_b_tile(kr), (short)0, s[t], false, false);
      s[t] = __builtin_amdgcn_wmma_f32_16x16x32_bf16(false, aq1, false,
               load_b_tile(kr + 32), (short)0, s[t], false, false);
    }

    const int4 mk = *(const int4*)&mrow[kc + 4 * ln];

#pragma unroll
    for (int r = 0; r < 8; ++r) {            // online softmax, row M = 8*lh + r
      float t0 = mk.x ? s[0][r] * ATT_SCALE : -1e9f;
      float t1 = mk.y ? s[1][r] * ATT_SCALE : -1e9f;
      float t2 = mk.z ? s[2][r] * ATT_SCALE : -1e9f;
      float t3 = mk.w ? s[3][r] * ATT_SCALE : -1e9f;
      float m2 = fmaxf(fmaxf(t0, t1), fmaxf(t2, t3));
      m2 = fmaxf(m2, __shfl_xor(m2, 1, 32));
      m2 = fmaxf(m2, __shfl_xor(m2, 2, 32));
      m2 = fmaxf(m2, __shfl_xor(m2, 4, 32));
      m2 = fmaxf(m2, __shfl_xor(m2, 8, 32));
      float mnew = fmaxf(vm[r], m2);
      float aold = __expf(vm[r] - mnew);
      float p0 = __expf(t0 - mnew), p1 = __expf(t1 - mnew);
      float p2 = __expf(t2 - mnew), p3 = __expf(t3 - mnew);
      float rs = (p0 + p1) + (p2 + p3);
      rs += __shfl_xor(rs, 1, 32);
      rs += __shfl_xor(rs, 2, 32);
      rs += __shfl_xor(rs, 4, 32);
      rs += __shfl_xor(rs, 8, 32);
      vl[r] = vl[r] * aold + rs;
      vm[r] = mnew;
#pragma unroll
      for (int nt = 0; nt < 4; ++nt) acc[nt][r] *= aold;
      bf16x4 pk;                             // packed b64 store, cols 4*ln..+3
      pk[0] = (__bf16)p0; pk[1] = (__bf16)p1; pk[2] = (__bf16)p2; pk[3] = (__bf16)p3;
      *(bf16x4*)&Pb[wave][(lh * 8 + r) * 64 + 4 * ln] = pk;
    }
    __syncthreads();                         // order Pb write -> A reads

    const __bf16* pr = &Pb[wave][ln * 64 + lh * 8];
    bf16x16 ap0 = load_a_tile(pr,      pr + 16);   // keys 0..31
    bf16x16 ap1 = load_a_tile(pr + 32, pr + 48);   // keys 32..63
#pragma unroll
    for (int nt = 0; nt < 4; ++nt) {         // O += P . V
      const __bf16* vr = &Vt[buf][(nt * 16 + ln) * 64 + lh * 8];
      acc[nt] = __builtin_amdgcn_wmma_f32_16x16x32_bf16(false, ap0, false,
                  load_b_tile(vr), (short)0, acc[nt], false, false);
      acc[nt] = __builtin_amdgcn_wmma_f32_16x16x32_bf16(false, ap1, false,
                  load_b_tile(vr + 32), (short)0, acc[nt], false, false);
    }
  }

#pragma unroll
  for (int nt = 0; nt < 4; ++nt) {
#pragma unroll
    for (int r = 0; r < 8; ++r) {
      int row = qt * 64 + wave * 16 + lh * 8 + r;
      int col = h * HD + nt * 16 + ln;
      out[((size_t)b * SEQ + row) * (NH * HD) + col] = acc[nt][r] / vl[r];
    }
  }
}

// ---------------------------------------------------------------------------
extern "C" void kernel_launch(void* const* d_in, const int* in_sizes, int n_in,
                              void* d_out, int out_size, void* d_ws, size_t ws_size,
                              hipStream_t stream) {
  const float* Q    = (const float*)d_in[0];
  const float* K    = (const float*)d_in[1];
  const float* V    = (const float*)d_in[2];
  const int*   mask = (const int*)  d_in[3];
  const float* Wq   = (const float*)d_in[4];
  const float* bq   = (const float*)d_in[5];
  const float* Wk   = (const float*)d_in[6];
  const float* bk   = (const float*)d_in[7];
  const float* Wv   = (const float*)d_in[8];
  const float* bv   = (const float*)d_in[9];
  float* out = (float*)d_out;

  const size_t per  = (size_t)BATCH * NH * SEQ * HD;    // 8M bf16 elements
  const size_t wsz  = (size_t)DMODEL * DMODEL;          // 1M bf16 elements
  __bf16* qws  = (__bf16*)d_ws;
  __bf16* kws  = qws + per;
  __bf16* vwsT = kws + per;
  __bf16* wtq  = vwsT + per;
  __bf16* wtk  = wtq + wsz;
  __bf16* wtv  = wtk + wsz;

  dim3 tgrid(DMODEL / 64, DMODEL / 64);                 // 16 x 16
  mha_wt_bf16<<<tgrid, 256, 0, stream>>>(Wq, wtq);
  mha_wt_bf16<<<tgrid, 256, 0, stream>>>(Wk, wtk);
  mha_wt_bf16<<<tgrid, 256, 0, stream>>>(Wv, wtv);

  dim3 pgrid(BATCH * SEQ / 64, DMODEL / 128);           // 128 x 8
  mha_proj_wmma<<<pgrid, 128, 0, stream>>>(Q, wtq, bq, qws, 0);
  mha_proj_wmma<<<pgrid, 128, 0, stream>>>(K, wtk, bk, kws, 0);
  mha_proj_wmma<<<pgrid, 128, 0, stream>>>(V, wtv, bv, vwsT, 1);

  dim3 agrid(SEQ / 64, NH, BATCH);                      // 32 x 16 x 4
  mha_flash_wmma<<<agrid, 128, 0, stream>>>(qws, kws, vwsT, mask, out);
}
